// ExpDock_36000415875251
// MI455X (gfx1250) — matrix operand order repaired and checked by reading, not compiled, not run on hardware
//
#include <hip/hip_runtime.h>

// ---------------- problem constants ----------------
#define Bc      2
#define NPART_  5000
#define N_      20000            // B * 2 * NPART
#define E_      180000           // 9 * N
#define HD_     128
#define NODE_IN_ 96
#define EDGE_IN_ 16
#define L_      4
#define KH_     10
#define NK_     50
#define EPS_    1e-3f
#define NSTD_   10.0f
#define EK_PAD  288              // 2*HD + EDGE_IN + 1 = 273 -> pad to 288 (mult of 32)

typedef __bf16 bf16;
typedef __attribute__((ext_vector_type(16))) bf16  v16bf;
typedef __attribute__((ext_vector_type(8)))  float v8f;
typedef unsigned int v4u __attribute__((ext_vector_type(4)));
typedef int v4i __attribute__((ext_vector_type(4)));
typedef int v8i __attribute__((ext_vector_type(8)));

struct U4 { unsigned int a, b, c, d; };
union AFrag { v16bf v; U4 u[2]; };

// =====================================================================
// WMMA GEMM:  C[M x 128] = act(A[M x Kpad] @ W[Kpad x 128] + bias) (+res)
// A row-major bf16 (lda = Kpad), W pre-packed into B-fragment layout and
// staged into LDS once per block via the Tensor Data Mover (TENSORcnt).
// One wave -> 32 rows x 128 cols (two M-tiles, 16 v8f accumulators);
// each LDS B-fragment feeds two WMMAs. Block = 8 waves = 256 rows.
// =====================================================================
__global__ __launch_bounds__(256) void k_gemm(
    const bf16* __restrict__ A, const bf16* __restrict__ Wp,
    const float* __restrict__ bias, const float* __restrict__ res,
    float* __restrict__ Cf, bf16* __restrict__ Cbf,
    int M, int Kpad, int act)
{
  extern __shared__ bf16 smemW[];          // Kpad*128 bf16
  int tid  = threadIdx.x;
  int wave = tid >> 5;
  int lane = tid & 31;

  // ---- stage packed weights into LDS (async tensor DMA, wave 0 only) ----
#if __has_builtin(__builtin_amdgcn_tensor_load_to_lds)
  if (tid < 32) {
    unsigned n8 = (unsigned)(Kpad * 32);                 // # of 8-byte elems
    unsigned long long ga = (unsigned long long)(size_t)Wp;
    v4u g0;
    g0[0] = 1u;                                          // count=1, user D#
    g0[1] = 0u;                                          // lds_addr = dynamic LDS base
    g0[2] = (unsigned)(ga & 0xffffffffull);              // global_addr[31:0]
    g0[3] = (unsigned)((ga >> 32) & 0x01ffffffull) | 0x80000000u; // [56:32] | type=2
    v8i g1;
    g1[0] = 0x00030000;                                  // data_size = 8 bytes
    g1[1] = (int)((n8 & 0xffffu) << 16);                 // tensor_dim0[15:0]
    g1[2] = (int)(((n8 >> 16) & 0xffffu) | (1u << 16));  // dim0[31:16] | tensor_dim1=1
    g1[3] = (int)((n8 & 0xffffu) << 16);                 // tile_dim0
    g1[4] = 1;                                           // tile_dim1 = 1
    g1[5] = (int)n8;                                     // tensor_dim0_stride
    g1[6] = 0;
    g1[7] = 0;
    v4i z4 = {0, 0, 0, 0};
#if defined(__clang_major__) && (__clang_major__ >= 23)
    v8i z8 = {0, 0, 0, 0, 0, 0, 0, 0};
    __builtin_amdgcn_tensor_load_to_lds(g0, g1, z4, z4, z8, 0);
#else
    __builtin_amdgcn_tensor_load_to_lds(g0, g1, z4, z4, 0);
#endif
    __builtin_amdgcn_s_wait_tensorcnt(0);
  }
#else
  for (int i = tid; i < Kpad * 32; i += 256)
    ((unsigned long long*)smemW)[i] = ((const unsigned long long*)Wp)[i];
#endif
  __syncthreads();

  int tm = blockIdx.x * 256 + wave * 32;     // wave-uniform tail guard (M % 32 == 0)
  if (tm < M) {
    int ktiles = Kpad >> 5;
    v8f acc0[8] = {}, acc1[8] = {};
    int lr = lane & 15;                      // A: row-in-tile, B: col-in-tile
    int hi = lane >> 4;                      // lane half
    const bf16* ar0 = A + (size_t)(tm + lr) * Kpad + hi * 16;
    const bf16* ar1 = A + (size_t)(tm + 16 + lr) * Kpad + hi * 16;
    for (int kt = 0; kt < ktiles; ++kt) {
      AFrag a0, a1;
      const bf16* p0 = ar0 + kt * 32;
      const bf16* p1 = ar1 + kt * 32;
      a0.u[0] = *(const U4*)(p0);            // K = {0..7}   (+16 if lane>=16)
      a0.u[1] = *(const U4*)(p0 + 8);        // K = {8..15}  (+16 if lane>=16)
      a1.u[0] = *(const U4*)(p1);
      a1.u[1] = *(const U4*)(p1 + 8);
      const bf16* wb = smemW + (size_t)(kt * 8) * 512 + lane * 16;
#pragma unroll
      for (int nt = 0; nt < 8; ++nt) {
        v16bf bw = *(const v16bf*)(wb + nt * 512);
        acc0[nt] = __builtin_amdgcn_wmma_f32_16x16x32_bf16(
            false, a0.v, false, bw, (short)0, acc0[nt], false, false);
        acc1[nt] = __builtin_amdgcn_wmma_f32_16x16x32_bf16(
            false, a1.v, false, bw, (short)0, acc1[nt], false, false);
      }
    }
    int rb0 = tm + hi * 8;
    int rb1 = tm + 16 + hi * 8;
#pragma unroll
    for (int nt = 0; nt < 8; ++nt) {
      int n = nt * 16 + lr;
      float bv = bias ? bias[n] : 0.f;
#pragma unroll
      for (int r = 0; r < 8; ++r) {
        float v0 = acc0[nt][r] + bv;
        if (act) v0 = v0 / (1.f + __expf(-v0));   // silu
        size_t i0 = (size_t)(rb0 + r) * 128 + n;
        if (res) v0 += res[i0];
        if (Cf)  Cf[i0]  = v0;
        if (Cbf) Cbf[i0] = (bf16)v0;
        float v1 = acc1[nt][r] + bv;
        if (act) v1 = v1 / (1.f + __expf(-v1));
        size_t i1 = (size_t)(rb1 + r) * 128 + n;
        if (res) v1 += res[i1];
        if (Cf)  Cf[i1]  = v1;
        if (Cbf) Cbf[i1] = (bf16)v1;
      }
    }
  }
}

// Pack f32 weight [Kreal x 128] -> bf16 B-fragments, zero-padded to Kpad.
// Fragment layout: frag = kt*8+nt ; lane<16: K=kt*32+i, lane>=16: K=kt*32+16+i
__global__ void k_pack(const float* __restrict__ W, bf16* __restrict__ Wp,
                       int Kreal, int Kpad)
{
  int idx = blockIdx.x * blockDim.x + threadIdx.x;   // (frag)*32 + lane
  int total = (Kpad >> 5) * 256;
  if (idx >= total) return;
  int lane = idx & 31;
  int frag = idx >> 5;
  int nt = frag & 7, kt = frag >> 3;
  int n  = nt * 16 + (lane & 15);
  int kb = kt * 32 + ((lane < 16) ? 0 : 16);
  bf16* outp = Wp + (size_t)idx * 16;
  for (int i = 0; i < 16; ++i) {
    int k = kb + i;
    outp[i] = (bf16)((k < Kreal) ? W[(size_t)k * 128 + n] : 0.f);
  }
}

__global__ void k_cast(const float* __restrict__ s, bf16* __restrict__ d, int n) {
  int i = blockIdx.x * blockDim.x + threadIdx.x;
  if (i < n) d[i] = (bf16)s[i];
}

// ---------------- preprocessing ----------------
__global__ void k_prep(const float* __restrict__ X, const float* __restrict__ center,
                       const float* __restrict__ rot, const float* __restrict__ trans,
                       const int* __restrict__ bid, const int* __restrict__ Seg,
                       float* __restrict__ Xt, float* __restrict__ ori,
                       float* __restrict__ ini)
{
  int i = blockIdx.x * blockDim.x + threadIdx.x;
  if (i >= N_) return;
  int b = bid[i];
  float x0 = (X[i*3+0] - center[b*3+0]) * (1.f / NSTD_);
  float x1 = (X[i*3+1] - center[b*3+1]) * (1.f / NSTD_);
  float x2 = (X[i*3+2] - center[b*3+2]) * (1.f / NSTD_);
  ori[i*3+0] = x0; ori[i*3+1] = x1; ori[i*3+2] = x2;
  float y0 = x0, y1 = x1, y2 = x2;
  if (Seg[i] == 0) {
    const float* r = rot + b*9; const float* t = trans + b*3;
    y0 = x0*r[0] + x1*r[3] + x2*r[6] + t[0];
    y1 = x0*r[1] + x1*r[4] + x2*r[7] + t[1];
    y2 = x0*r[2] + x1*r[5] + x2*r[8] + t[2];
  }
  Xt[i*3+0] = y0; Xt[i*3+1] = y1; Xt[i*3+2] = y2;
  ini[i*3+0] = y0; ini[i*3+1] = y1; ini[i*3+2] = y2;
}

__global__ void k_prep_kp(const float* __restrict__ keypoints, const float* __restrict__ center,
                          const float* __restrict__ rot, const float* __restrict__ trans,
                          const int* __restrict__ k_bid,
                          float* __restrict__ kp, float* __restrict__ tkp)
{
  int i = blockIdx.x * blockDim.x + threadIdx.x;
  if (i >= Bc * NK_) return;
  int b = k_bid[i];
  float x0 = (keypoints[i*3+0] - center[b*3+0]) * (1.f / NSTD_);
  float x1 = (keypoints[i*3+1] - center[b*3+1]) * (1.f / NSTD_);
  float x2 = (keypoints[i*3+2] - center[b*3+2]) * (1.f / NSTD_);
  kp[i*3+0] = x0; kp[i*3+1] = x1; kp[i*3+2] = x2;
  const float* r = rot + b*9; const float* t = trans + b*3;
  tkp[i*3+0] = x0*r[0] + x1*r[3] + x2*r[6] + t[0];
  tkp[i*3+1] = x0*r[1] + x1*r[4] + x2*r[7] + t[1];
  tkp[i*3+2] = x0*r[2] + x1*r[5] + x2*r[8] + t[2];
}

__global__ void k_nv0(const float* __restrict__ Xt, const int* __restrict__ Seg,
                      float* __restrict__ nv)
{
  int i = blockIdx.x * blockDim.x + threadIdx.x;
  if (i >= N_) return;
  float s0 = 0.f, s1 = 0.f, s2 = 0.f;
  if (i + 1 < N_ && Seg[i+1] == Seg[i]) { s0 = Xt[(i+1)*3+0]; s1 = Xt[(i+1)*3+1]; s2 = Xt[(i+1)*3+2]; }
  float d0 = s0 - Xt[i*3+0], d1 = s1 - Xt[i*3+1], d2 = s2 - Xt[i*3+2];
  float nr = sqrtf(d0*d0 + d1*d1 + d2*d2) + EPS_;
  nv[i*3+0] = d0/nr; nv[i*3+1] = d1/nr; nv[i*3+2] = d2/nr;
}

// ---------------- per-layer kernels ----------------
__global__ void k_edge_build(const int* __restrict__ edges, const float* __restrict__ Xt,
                             const bf16* __restrict__ Hbf, const float* __restrict__ edge_attr,
                             bf16* __restrict__ Ab, float* __restrict__ cdb)
{
  int e = blockIdx.x * blockDim.x + threadIdx.x;
  if (e >= E_) return;
  int r = edges[e], c = edges[E_ + e];
  float c0 = Xt[r*3+0] - Xt[c*3+0];
  float c1 = Xt[r*3+1] - Xt[c*3+1];
  float c2 = Xt[r*3+2] - Xt[c*3+2];
  cdb[e*3+0] = c0; cdb[e*3+1] = c1; cdb[e*3+2] = c2;
  float d2 = c0*c0 + c1*c1 + c2*c2;
  bf16* outp = Ab + (size_t)e * EK_PAD;
  U4* o = (U4*)outp;
  const U4* hr = (const U4*)(Hbf + (size_t)r * 128);
  const U4* hc = (const U4*)(Hbf + (size_t)c * 128);
#pragma unroll
  for (int i = 0; i < 16; ++i) o[i] = hr[i];
#pragma unroll
  for (int i = 0; i < 16; ++i) o[16 + i] = hc[i];
  outp[256] = (bf16)d2;
  for (int j = 0; j < 16; ++j) outp[257 + j] = (bf16)edge_attr[(size_t)e*16 + j];
  for (int j = 273; j < EK_PAD; ++j) outp[j] = (bf16)0.f;
}

// wave per edge: mx = m.wx, mn = m.wn (shuffle reduce), scatter atomics
__global__ __launch_bounds__(256) void k_edge_scatter(
    const int* __restrict__ edges, const float* __restrict__ cdb,
    const float* __restrict__ m, const float* __restrict__ wx,
    const float* __restrict__ wn, float* __restrict__ Xacc,
    float* __restrict__ nvacc, float* __restrict__ msum, float* __restrict__ cnt)
{
  int wave = threadIdx.x >> 5, lane = threadIdx.x & 31;
  int e = blockIdx.x * 8 + wave;
  if (e >= E_) return;
  const float* me = m + (size_t)e * 128;
  float v[4]; float px = 0.f, pn = 0.f;
#pragma unroll
  for (int j = 0; j < 4; ++j) {
    v[j] = me[lane + 32*j];
    px += v[j] * wx[lane + 32*j];
    pn += v[j] * wn[lane + 32*j];
  }
  for (int off = 16; off; off >>= 1) {
    px += __shfl_xor(px, off, 32);
    pn += __shfl_xor(pn, off, 32);
  }
  int r = edges[e];
#pragma unroll
  for (int j = 0; j < 4; ++j) atomicAdd(&msum[(size_t)r*128 + lane + 32*j], v[j]);
  if (lane < 3) {
    float cd = cdb[e*3 + lane];
    atomicAdd(&Xacc[r*3 + lane],  cd * px);
    atomicAdd(&nvacc[r*3 + lane], cd * pn);
  } else if (lane == 3) {
    atomicAdd(&cnt[r], 1.f);
  }
}

// per-node update + build node-GEMM input [H | segmean(m)] (bf16)
__global__ void k_node_update(float* __restrict__ Xt, float* __restrict__ nv,
                              const float* __restrict__ Xacc, const float* __restrict__ nvacc,
                              const float* __restrict__ msum, const float* __restrict__ cnt,
                              const bf16* __restrict__ Hbf, bf16* __restrict__ Nin)
{
  int n = blockIdx.x, j = threadIdx.x;
  float c = fmaxf(cnt[n], 1.f);
  Nin[(size_t)n*256 + j]       = Hbf[(size_t)n*128 + j];
  Nin[(size_t)n*256 + 128 + j] = (bf16)(msum[(size_t)n*128 + j] / c);
  if (j < 3) Xt[n*3 + j] += Xacc[n*3 + j] / c;
  if (j == 3) {
    float u0 = nv[n*3+0] + nvacc[n*3+0] / c;
    float u1 = nv[n*3+1] + nvacc[n*3+1] / c;
    float u2 = nv[n*3+2] + nvacc[n*3+2] / c;
    float nr = sqrtf(u0*u0 + u1*u1 + u2*u2) + EPS_;
    nv[n*3+0] = u0/nr; nv[n*3+1] = u1/nr; nv[n*3+2] = u2/nr;
  }
}

__global__ void k_fb_edge(const int* __restrict__ edges, const float* __restrict__ Xt,
                          const float* __restrict__ nv, float* __restrict__ fb_node)
{
  int e = blockIdx.x * blockDim.x + threadIdx.x;
  if (e >= E_) return;
  int r = edges[e], c = edges[E_ + e];
  float c0 = Xt[r*3+0] - Xt[c*3+0];
  float c1 = Xt[r*3+1] - Xt[c*3+1];
  float c2 = Xt[r*3+2] - Xt[c*3+2];
  float prod = c0*nv[r*3+0] + c1*nv[r*3+1] + c2*nv[r*3+2];
  float nrm = sqrtf(c0*c0 + c1*c1 + c2*c2);
  nrm = fminf(fmaxf(nrm, 1e-3f), 2.f);
  atomicAdd(&fb_node[r], prod / (nrm*nrm*nrm));
}

__global__ void k_fb_node(const float* __restrict__ fb_node, const float* __restrict__ cnt,
                          const int* __restrict__ bid, float* __restrict__ fb_bid)
{
  int n = blockIdx.x * blockDim.x + threadIdx.x;
  if (n >= N_) return;
  float c = fmaxf(cnt[n], 1.f);
  float fbn = sqrtf(fabsf(fb_node[n] / c) + EPS_);
  atomicAdd(&fb_bid[bid[n]], fbn);
}

__global__ void k_fb_final(const float* __restrict__ fb_bid, float* __restrict__ acc)
{
  float s = 0.f;
  for (int b = 0; b < Bc; ++b) s += fb_bid[b] / (2.f * (float)NPART_);
  acc[0] += s / (float)Bc;
}

// ---------------- keypoint attention ----------------
__global__ void k_hmean(const float* __restrict__ H, float* __restrict__ Hm)
{
  int bs = blockIdx.x, j = threadIdx.x;     // bs in [0,4): (b,seg)
  const float* Hp = H + (size_t)bs * NPART_ * HD_;
  float s = 0.f;
  for (int n = 0; n < NPART_; ++n) s += Hp[(size_t)n*HD_ + j];
  Hm[bs*HD_ + j] = s / (float)NPART_;
}

__global__ void k_qvec(const float* __restrict__ W1k, const float* __restrict__ W2k,
                       const float* __restrict__ Hm, float* __restrict__ qv)
{
  int blk = blockIdx.x, d = threadIdx.x;    // blk = (b*2+side)*K + k
  int k = blk % KH_; int side = (blk / KH_) % 2; int b = blk / (2*KH_);
  const float* Wk = (side == 0 ? W1k : W2k) + (size_t)k * HD_ * HD_;
  const float* hm = Hm + (size_t)(b*2 + (side == 0 ? 1 : 0)) * HD_;
  float s = 0.f;
  for (int e = 0; e < HD_; ++e) s += Wk[(size_t)d*HD_ + e] * hm[e];
  qv[(size_t)blk*HD_ + d] = s;
}

__global__ __launch_bounds__(256) void k_attn_logits(const float* __restrict__ H,
                                                     const float* __restrict__ qv,
                                                     float* __restrict__ lg)
{
  __shared__ float sred[256];
  int blk = blockIdx.x, tid = threadIdx.x;
  int side = (blk / KH_) % 2; int b = blk / (2*KH_);
  const float* Hs = H + (size_t)(b*2 + side) * NPART_ * HD_;
  const float* q  = qv + (size_t)blk * HD_;
  float* l = lg + (size_t)blk * NPART_;
  const float scale = 0.08838834764831845f;   // 1/sqrt(128)
  float mx = -1e30f;
  for (int n = tid; n < NPART_; n += 256) {
    float s = 0.f;
    for (int j = 0; j < HD_; ++j) s += Hs[(size_t)n*HD_ + j] * q[j];
    s *= scale; l[n] = s; mx = fmaxf(mx, s);
  }
  sred[tid] = mx; __syncthreads();
  for (int s = 128; s; s >>= 1) { if (tid < s) sred[tid] = fmaxf(sred[tid], sred[tid+s]); __syncthreads(); }
  float gmx = sred[0]; __syncthreads();
  float se = 0.f;
  for (int n = tid; n < NPART_; n += 256) se += __expf(l[n] - gmx);
  sred[tid] = se; __syncthreads();
  for (int s = 128; s; s >>= 1) { if (tid < s) sred[tid] += sred[tid+s]; __syncthreads(); }
  float gse = sred[0];
  for (int n = tid; n < NPART_; n += 256) l[n] = __expf(l[n] - gmx) / gse;
}

__global__ void k_attn_apply(const float* __restrict__ H, const float* __restrict__ Xt,
                             const float* __restrict__ lg, float* __restrict__ Y,
                             float* __restrict__ YH)
{
  int blk = blockIdx.x, j = threadIdx.x;
  int side = (blk / KH_) % 2; int b = blk / (2*KH_);
  const float* Hs = H  + (size_t)(b*2 + side) * NPART_ * HD_;
  const float* Xs = Xt + (size_t)(b*2 + side) * NPART_ * 3;
  const float* l  = lg + (size_t)blk * NPART_;
  float s = 0.f;
  for (int n = 0; n < NPART_; ++n) s += l[n] * Hs[(size_t)n*HD_ + j];
  YH[(size_t)blk*HD_ + j] = s;
  if (j < 3) {
    float y = 0.f;
    for (int n = 0; n < NPART_; ++n) y += l[n] * Xs[n*3 + j];
    Y[(size_t)blk*3 + j] = y;
  }
}

// ---------------- Kabsch / final losses ----------------
__device__ inline float det3(const float M[3][3]) {
  return M[0][0]*(M[1][1]*M[2][2]-M[1][2]*M[2][1])
       - M[0][1]*(M[1][0]*M[2][2]-M[1][2]*M[2][0])
       + M[0][2]*(M[1][0]*M[2][1]-M[1][1]*M[2][0]);
}

__device__ inline void jacobi3(float A[3][3], float V[3][3]) {
  for (int i = 0; i < 3; ++i) for (int j = 0; j < 3; ++j) V[i][j] = (i==j) ? 1.f : 0.f;
  for (int sweep = 0; sweep < 16; ++sweep)
    for (int p = 0; p < 2; ++p)
      for (int q = p+1; q < 3; ++q) {
        float apq = A[p][q];
        if (fabsf(apq) < 1e-12f) continue;
        float theta = 0.5f * (A[q][q] - A[p][p]) / apq;
        float t = ((theta >= 0.f) ? 1.f : -1.f) / (fabsf(theta) + sqrtf(theta*theta + 1.f));
        float c = rsqrtf(t*t + 1.f);
        float s = t * c;
        for (int i = 0; i < 3; ++i) { float aip=A[i][p], aiq=A[i][q]; A[i][p]=c*aip-s*aiq; A[i][q]=s*aip+c*aiq; }
        for (int i = 0; i < 3; ++i) { float api=A[p][i], aqi=A[q][i]; A[p][i]=c*api-s*aqi; A[q][i]=s*api+c*aqi; }
        for (int i = 0; i < 3; ++i) { float vip=V[i][p], viq=V[i][q]; V[i][p]=c*vip-s*viq; V[i][q]=s*vip+c*viq; }
      }
}

__global__ __launch_bounds__(256) void k_final(
    const float* __restrict__ Yw, const float* __restrict__ YHw,
    const float* __restrict__ kp, const float* __restrict__ tkp,
    const float* __restrict__ rot, const float* __restrict__ trans,
    const float* __restrict__ iniX, const float* __restrict__ oriX,
    float* __restrict__ acc)
{
  __shared__ float sR[3][3], sT[3], sS[3], sred[256];
  int b = blockIdx.x, tid = threadIdx.x;
  if (tid == 0) {
    const float* Y1  = Yw  + (size_t)(b*2 + 0) * KH_ * 3;
    const float* Y2  = Yw  + (size_t)(b*2 + 1) * KH_ * 3;
    const float* YH1 = YHw + (size_t)(b*2 + 0) * KH_ * HD_;
    const float* YH2 = YHw + (size_t)(b*2 + 1) * KH_ * HD_;
    const float* P1 = tkp + (size_t)b * NK_ * 3;
    const float* P2 = kp  + (size_t)b * NK_ * 3;
    int mi1[KH_], mi2[KH_];
    float ot = 0.f;
    for (int k = 0; k < KH_; ++k) {
      float best = 1e30f; int bi = 0;
      for (int j = 0; j < NK_; ++j) {
        float d0=Y1[k*3+0]-P1[j*3+0], d1=Y1[k*3+1]-P1[j*3+1], d2=Y1[k*3+2]-P1[j*3+2];
        float dd = d0*d0 + d1*d1 + d2*d2;
        if (dd < best) { best = dd; bi = j; }
      }
      mi1[k] = bi; ot += best;
    }
    for (int k = 0; k < KH_; ++k) {
      float best = 1e30f; int bi = 0;
      for (int j = 0; j < NK_; ++j) {
        float d0=Y2[k*3+0]-P2[j*3+0], d1=Y2[k*3+1]-P2[j*3+1], d2=Y2[k*3+2]-P2[j*3+2];
        float dd = d0*d0 + d1*d1 + d2*d2;
        if (dd < best) { best = dd; bi = j; }
      }
      mi2[k] = bi; ot += best;
    }
    ot /= (float)(KH_ * 3);
    // Kabsch(Y1 -> Y2)
    float ma[3] = {0,0,0}, mb[3] = {0,0,0};
    for (int k = 0; k < KH_; ++k)
      for (int d = 0; d < 3; ++d) { ma[d] += Y1[k*3+d] / KH_; mb[d] += Y2[k*3+d] / KH_; }
    float C[3][3] = {{0,0,0},{0,0,0},{0,0,0}};
    for (int k = 0; k < KH_; ++k)
      for (int d = 0; d < 3; ++d)
        for (int e = 0; e < 3; ++e)
          C[d][e] += (Y1[k*3+d] - ma[d]) * (Y2[k*3+e] - mb[e]);
    float S[3][3], V[3][3];
    for (int i = 0; i < 3; ++i)
      for (int j = 0; j < 3; ++j)
        S[i][j] = C[0][i]*C[0][j] + C[1][i]*C[1][j] + C[2][i]*C[2][j];
    jacobi3(S, V);
    float w[3] = { S[0][0], S[1][1], S[2][2] };
    for (int i = 0; i < 2; ++i)
      for (int j = i+1; j < 3; ++j)
        if (w[j] > w[i]) {
          float t0 = w[i]; w[i] = w[j]; w[j] = t0;
          for (int d = 0; d < 3; ++d) { float tv = V[d][i]; V[d][i] = V[d][j]; V[d][j] = tv; }
        }
    float U[3][3];
    for (int i = 0; i < 3; ++i) {
      float s = sqrtf(fmaxf(w[i], 0.f));
      float inv = 1.f / fmaxf(s, 1e-12f);
      for (int d = 0; d < 3; ++d)
        U[d][i] = (C[d][0]*V[0][i] + C[d][1]*V[1][i] + C[d][2]*V[2][i]) * inv;
    }
    float dsgn = (det3(U) * det3(V) >= 0.f) ? 1.f : -1.f;
    float R[3][3];
    for (int d = 0; d < 3; ++d)
      for (int e = 0; e < 3; ++e)
        R[d][e] = U[d][0]*V[e][0] + U[d][1]*V[e][1] + dsgn*U[d][2]*V[e][2];
    float tt[3];
    for (int e = 0; e < 3; ++e)
      tt[e] = mb[e] - (ma[0]*R[0][e] + ma[1]*R[1][e] + ma[2]*R[2][e]);
    // dock
    const float* rb = rot + b*9; const float* tb = trans + b*3;
    float dock = 0.f;
    for (int i = 0; i < 3; ++i)
      for (int j = 0; j < 3; ++j) {
        float v = rb[i*3+0]*R[0][j] + rb[i*3+1]*R[1][j] + rb[i*3+2]*R[2][j] - ((i==j)?1.f:0.f);
        dock += v*v;
      }
    dock /= 9.f;
    float tv2 = 0.f;
    for (int e = 0; e < 3; ++e) {
      float v = tb[0]*R[0][e] + tb[1]*R[1][e] + tb[2]*R[2][e] + tt[e];
      tv2 += v*v;
    }
    dock += tv2 / 3.f;
    // match
    const float log1pe = logf(1.f + EPS_);
    float match = 0.f;
    for (int k = 0; k < KH_; ++k) {   // z1: D12 = d2(P2[mi1[k]], Y2[j])
      float bmax = -1e30f, bmin = 1e30f; int amax = 0, amin = 0;
      const float* p = P2 + mi1[k]*3;
      for (int j = 0; j < KH_; ++j) {
        float d0=p[0]-Y2[j*3+0], d1=p[1]-Y2[j*3+1], d2=p[2]-Y2[j*3+2];
        float dd = d0*d0 + d1*d1 + d2*d2;
        if (dd > bmax) { bmax = dd; amax = j; }
        if (dd < bmin) { bmin = dd; amin = j; }
      }
      float dpx = 0.f, dpn = 0.f;
      for (int j2 = 0; j2 < HD_; ++j2) {
        dpx += YH1[k*HD_+j2]*YH2[amax*HD_+j2];
        dpn += YH1[k*HD_+j2]*YH2[amin*HD_+j2];
      }
      float z = 0.8f*dpx - dpn;
      float m0 = fmaxf(log1pe, z);
      match += m0 + log1pf(__expf(fminf(log1pe, z) - m0));
    }
    for (int k = 0; k < KH_; ++k) {   // z2: D21 = d2(P1[mi2[k]], Y1[j])
      float bmax = -1e30f, bmin = 1e30f; int amax = 0, amin = 0;
      const float* p = P1 + mi2[k]*3;
      for (int j = 0; j < KH_; ++j) {
        float d0=p[0]-Y1[j*3+0], d1=p[1]-Y1[j*3+1], d2=p[2]-Y1[j*3+2];
        float dd = d0*d0 + d1*d1 + d2*d2;
        if (dd > bmax) { bmax = dd; amax = j; }
        if (dd < bmin) { bmin = dd; amin = j; }
      }
      float dpx = 0.f, dpn = 0.f;
      for (int j2 = 0; j2 < HD_; ++j2) {
        dpx += YH2[k*HD_+j2]*YH1[amax*HD_+j2];
        dpn += YH2[k*HD_+j2]*YH1[amin*HD_+j2];
      }
      float z = 0.8f*dpx - dpn;
      float m0 = fmaxf(log1pe, z);
      match += m0 + log1pf(__expf(fminf(log1pe, z) - m0));
    }
    match = match / (float)KH_ / (float)KH_;
    for (int d = 0; d < 3; ++d) { for (int e = 0; e < 3; ++e) sR[d][e] = R[d][e]; sT[d] = tt[d]; }
    sS[0] = ot; sS[1] = dock; sS[2] = match;
  }
  __syncthreads();
  // rmsd over ligand part
  float part = 0.f;
  const float* iX = iniX + (size_t)(b*2) * NPART_ * 3;
  const float* oX = oriX + (size_t)(b*2) * NPART_ * 3;
  for (int n = tid; n < NPART_; n += 256) {
    float p0 = iX[n*3+0], p1 = iX[n*3+1], p2 = iX[n*3+2];
    for (int e = 0; e < 3; ++e) {
      float v = p0*sR[0][e] + p1*sR[1][e] + p2*sR[2][e] + sT[e] - oX[n*3+e];
      part += v*v;
    }
  }
  sred[tid] = part; __syncthreads();
  for (int s = 128; s; s >>= 1) { if (tid < s) sred[tid] += sred[tid+s]; __syncthreads(); }
  if (tid == 0) {
    atomicAdd(&acc[1], sS[0] / (float)Bc);
    atomicAdd(&acc[2], sS[1] / (float)Bc);
    atomicAdd(&acc[3], sS[2] / (float)Bc);
    atomicAdd(&acc[4], sred[0] / ((float)NPART_ * 3.f) / (float)Bc);
  }
}

__global__ void k_loss_final(const float* __restrict__ acc, float* __restrict__ out)
{
  float fb_loss = acc[0] / (float)L_;
  float ot = acc[1], dock = acc[2], match = acc[3], rmsd = acc[4];
  out[0] = fb_loss + ot + 5.f*match + dock + rmsd;
  out[1] = fb_loss; out[2] = ot; out[3] = dock; out[4] = match; out[5] = rmsd;
}

// =====================================================================
extern "C" void kernel_launch(void* const* d_in, const int* in_sizes, int n_in,
                              void* d_out, int out_size, void* d_ws, size_t ws_size,
                              hipStream_t stream)
{
  (void)in_sizes; (void)n_in; (void)out_size; (void)ws_size;
  const float* X         = (const float*)d_in[0];
  const float* center    = (const float*)d_in[1];
  const float* keypoints = (const float*)d_in[2];
  const float* rot       = (const float*)d_in[3];
  const float* trans     = (const float*)d_in[4];
  const float* node_attr = (const float*)d_in[5];
  const float* edge_attr = (const float*)d_in[6];
  const float* W_in      = (const float*)d_in[7];
  const float* b_in      = (const float*)d_in[8];
  const float* We1       = (const float*)d_in[9];
  const float* be1       = (const float*)d_in[10];
  const float* We2       = (const float*)d_in[11];
  const float* be2       = (const float*)d_in[12];
  const float* Wx        = (const float*)d_in[13];
  const float* Wn        = (const float*)d_in[14];
  const float* Wh        = (const float*)d_in[15];
  const float* bh        = (const float*)d_in[16];
  const float* W1k       = (const float*)d_in[17];
  const float* W2k       = (const float*)d_in[18];
  const int*   edges     = (const int*)d_in[19];
  const int*   bid       = (const int*)d_in[20];
  const int*   Seg       = (const int*)d_in[21];
  const int*   k_bid     = (const int*)d_in[22];
  float* out = (float*)d_out;

  // ------- workspace bump allocator -------
  char* p = (char*)d_ws;
  auto alloc = [&](size_t bytes) -> void* {
    void* r = (void*)p; p += (bytes + 255) & ~(size_t)255; return r;
  };
  float* Xt      = (float*)alloc((size_t)N_*3*4);
  float* nv      = (float*)alloc((size_t)N_*3*4);
  float* oriX    = (float*)alloc((size_t)N_*3*4);
  float* iniX    = (float*)alloc((size_t)N_*3*4);
  float* kpw     = (float*)alloc((size_t)Bc*NK_*3*4);
  float* tkpw    = (float*)alloc((size_t)Bc*NK_*3*4);
  float* H       = (float*)alloc((size_t)N_*HD_*4);
  float* mbuf    = (float*)alloc((size_t)E_*HD_*4);
  float* cdbuf   = (float*)alloc((size_t)E_*3*4);
  float* Xacc    = (float*)alloc((size_t)N_*3*4);
  float* nvacc   = (float*)alloc((size_t)N_*3*4);
  float* cnt     = (float*)alloc((size_t)N_*4);
  float* msum    = (float*)alloc((size_t)N_*HD_*4);
  float* fb_node = (float*)alloc((size_t)N_*4);
  float* fb_bid  = (float*)alloc((size_t)Bc*4);
  float* Hm      = (float*)alloc((size_t)Bc*2*HD_*4);
  float* qv      = (float*)alloc((size_t)Bc*2*KH_*HD_*4);
  float* logitsw = (float*)alloc((size_t)Bc*2*KH_*NPART_*4);
  float* Yw      = (float*)alloc((size_t)Bc*2*KH_*3*4);
  float* YHw     = (float*)alloc((size_t)Bc*2*KH_*HD_*4);
  float* acc     = (float*)alloc(8*4);
  bf16* Hbf   = (bf16*)alloc((size_t)N_*HD_*2);
  bf16* nabf  = (bf16*)alloc((size_t)N_*NODE_IN_*2);
  bf16* Abuf  = (bf16*)alloc((size_t)E_*EK_PAD*2);   // reused as node-GEMM input
  bf16* m1buf = (bf16*)alloc((size_t)E_*HD_*2);
  bf16* WinP  = (bf16*)alloc((size_t)NODE_IN_*HD_*2);
  bf16* We1P  = (bf16*)alloc((size_t)L_*EK_PAD*HD_*2);
  bf16* We2P  = (bf16*)alloc((size_t)L_*HD_*HD_*2);
  bf16* WhP   = (bf16*)alloc((size_t)L_*2*HD_*HD_*2);

  hipMemsetAsync(acc, 0, 8*4, stream);

  // ------- preprocessing -------
  k_prep<<<(N_+255)/256, 256, 0, stream>>>(X, center, rot, trans, bid, Seg, Xt, oriX, iniX);
  k_prep_kp<<<1, Bc*NK_, 0, stream>>>(keypoints, center, rot, trans, k_bid, kpw, tkpw);
  k_nv0<<<(N_+255)/256, 256, 0, stream>>>(Xt, Seg, nv);
  k_cast<<<((N_*NODE_IN_)+255)/256, 256, 0, stream>>>(node_attr, nabf, N_*NODE_IN_);

  // ------- weight packing (B-fragment layout) -------
  {
    int t0 = (NODE_IN_ >> 5) * 256;
    k_pack<<<(t0+255)/256, 256, 0, stream>>>(W_in, WinP, NODE_IN_, NODE_IN_);
    int t1 = (EK_PAD >> 5) * 256;
    int t2 = (HD_ >> 5) * 256;
    int t3 = ((2*HD_) >> 5) * 256;
    for (int l = 0; l < L_; ++l) {
      k_pack<<<(t1+255)/256, 256, 0, stream>>>(We1 + (size_t)l*273*HD_, We1P + (size_t)l*EK_PAD*HD_, 273, EK_PAD);
      k_pack<<<(t2+255)/256, 256, 0, stream>>>(We2 + (size_t)l*HD_*HD_, We2P + (size_t)l*HD_*HD_, HD_, HD_);
      k_pack<<<(t3+255)/256, 256, 0, stream>>>(Wh  + (size_t)l*2*HD_*HD_, WhP + (size_t)l*2*HD_*HD_, 2*HD_, 2*HD_);
    }
  }

  // ------- GEMM grids / LDS sizes -------
  int gN = (N_ + 255) / 256;                 // 79 blocks (M % 32 == 0)
  int gE = (E_ + 255) / 256;                 // 704 blocks
  size_t sh_in = (size_t)NODE_IN_ * 128 * 2; // 24 KB
  size_t sh_e1 = (size_t)EK_PAD  * 128 * 2;  // 72 KB
  size_t sh_e2 = (size_t)HD_     * 128 * 2;  // 32 KB
  size_t sh_h  = (size_t)2*HD_   * 128 * 2;  // 64 KB

  // ------- H0 = node_attr @ W_in + b_in -------
  k_gemm<<<gN, 256, sh_in, stream>>>(nabf, WinP, b_in, nullptr, H, Hbf, N_, NODE_IN_, 0);

  // ------- GNN layers -------
  for (int l = 0; l < L_; ++l) {
    hipMemsetAsync(Xacc,    0, (size_t)N_*3*4,   stream);
    hipMemsetAsync(nvacc,   0, (size_t)N_*3*4,   stream);
    hipMemsetAsync(cnt,     0, (size_t)N_*4,     stream);
    hipMemsetAsync(msum,    0, (size_t)N_*HD_*4, stream);
    hipMemsetAsync(fb_node, 0, (size_t)N_*4,     stream);
    hipMemsetAsync(fb_bid,  0, (size_t)Bc*4,     stream);

    k_edge_build<<<(E_+255)/256, 256, 0, stream>>>(edges, Xt, Hbf, edge_attr, Abuf, cdbuf);
    k_gemm<<<gE, 256, sh_e1, stream>>>(Abuf, We1P + (size_t)l*EK_PAD*HD_, be1 + l*HD_,
                                       nullptr, nullptr, m1buf, E_, EK_PAD, 1);
    k_gemm<<<gE, 256, sh_e2, stream>>>(m1buf, We2P + (size_t)l*HD_*HD_, be2 + l*HD_,
                                       nullptr, mbuf, nullptr, E_, HD_, 1);
    k_edge_scatter<<<E_/8, 256, 0, stream>>>(edges, cdbuf, mbuf, Wx + l*HD_, Wn + l*HD_,
                                             Xacc, nvacc, msum, cnt);
    k_node_update<<<N_, 128, 0, stream>>>(Xt, nv, Xacc, nvacc, msum, cnt, Hbf, Abuf);
    k_gemm<<<gN, 256, sh_h, stream>>>(Abuf, WhP + (size_t)l*2*HD_*HD_, bh + l*HD_,
                                      H, H, Hbf, N_, 2*HD_, 1);
    k_fb_edge<<<(E_+255)/256, 256, 0, stream>>>(edges, Xt, nv, fb_node);
    k_fb_node<<<(N_+255)/256, 256, 0, stream>>>(fb_node, cnt, bid, fb_bid);
    k_fb_final<<<1, 1, 0, stream>>>(fb_bid, acc);
  }

  // ------- keypoint attention + losses -------
  k_hmean<<<Bc*2, HD_, 0, stream>>>(H, Hm);
  k_qvec<<<Bc*2*KH_, HD_, 0, stream>>>(W1k, W2k, Hm, qv);
  k_attn_logits<<<Bc*2*KH_, 256, 0, stream>>>(H, qv, logitsw);
  k_attn_apply<<<Bc*2*KH_, HD_, 0, stream>>>(H, Xt, logitsw, Yw, YHw);
  k_final<<<Bc, 256, 0, stream>>>(Yw, YHw, kpw, tkpw, rot, trans, iniX, oriX, acc);
  k_loss_final<<<1, 1, 0, stream>>>(acc, out);
}
